// MF_10161892622937
// MI455X (gfx1250) — matrix operand order repaired and checked by reading, not compiled, not run on hardware
//
#include <hip/hip_runtime.h>

// MI455X / gfx1250 fused pair-scoring kernel, round 2.
// - One-shot staging kernel converts W0/W1 to f16 in WMMA B-fragment order (d_ws).
// - Main kernel: block = 16 pairs, 128 threads = 4 wave32.
//   gather->LDS, embeddings (VALU) written as f16 A-fragments, WMMA decoder
//   with b128 LDS fragment loads (no per-fragment cvt / scattered b32 loads).

typedef __attribute__((ext_vector_type(16))) _Float16 v16h;
typedef __attribute__((ext_vector_type(8)))  float    v8f;

#define S_      4
#define N_      16384
#define K_      128
#define PAIRS   16
#define THREADS 128

#define W0_ELEMS (256 * 128)          // 32768
#define W1_ELEMS (128 * 64)           // 8192
#define WST_HALFS (W0_ELEMS + W1_ELEMS)

union CFrag { v8f v; float f[8]; };

__device__ __forceinline__ float lrelu(float x) { return x > 0.f ? x : 0.01f * x; }

// ---------------------------------------------------------------------------
// One-shot: W0 [256x128] and W1 [128x64] f32 -> f16 in B-fragment order.
// B layout (32x16 f16 tile): lane L holds col N=L&15; half j holds K=8*(L>>4)+(j&7)+16*(j>>3).
// Fragment (kstep, ntile) stored as 32 lanes x 16 contiguous halfs.
// ---------------------------------------------------------------------------
__global__ void __launch_bounds__(256)
MF_stage_weights(const float* __restrict__ W0, const float* __restrict__ W1,
                 _Float16* __restrict__ wst)
{
    int g = blockIdx.x * 256 + threadIdx.x;
    if (g < W0_ELEMS) {                       // W0[k*128+n], 8 kstep x 8 ntile
        int i = g, n = i & 127, k = i >> 7;
        int kstep = k >> 5, kr = k & 31;
        int L = (n & 15) + ((kr & 8) << 1);
        wst[((kstep * 8 + (n >> 4)) * 32 + L) * 16 + (kr & 7) + ((kr & 16) >> 1)] = (_Float16)W0[i];
    } else if (g < WST_HALFS) {               // W1[k*64+n],  4 kstep x 4 ntile
        int i = g - W0_ELEMS, n = i & 63, k = i >> 6;
        int kstep = k >> 5, kr = k & 31;
        int L = (n & 15) + ((kr & 8) << 1);
        wst[W0_ELEMS + ((kstep * 4 + (n >> 4)) * 32 + L) * 16 + (kr & 7) + ((kr & 16) >> 1)] = (_Float16)W1[i];
    }
}

// ---------------------------------------------------------------------------
// Main fused kernel.
// ---------------------------------------------------------------------------
__global__ void __launch_bounds__(THREADS)
MF_pairmlp_wmma(const float* __restrict__ X,   const float* __restrict__ Y,
                const float* __restrict__ xatt,const float* __restrict__ yatt,
                const float* __restrict__ WcX1,const float* __restrict__ bcX1,
                const float* __restrict__ WcX2,const float* __restrict__ bcX2,
                const float* __restrict__ WcY1,const float* __restrict__ bcY1,
                const float* __restrict__ WcY2,const float* __restrict__ bcY2,
                const float* __restrict__ b0,  const float* __restrict__ b1,
                const float* __restrict__ W2,  const float* __restrict__ b2,
                const _Float16* __restrict__ wsrc,
                const int*   __restrict__ idx0,const int*   __restrict__ idx1,
                float* __restrict__ out)
{
    extern __shared__ char smem[];
    float*    Zf    = (float*)smem;                   // 16*2*4*128 f32 = 64KB
    _Float16* Wst   = (_Float16*)(smem + 65536);      // 40960 halfs    = 80KB
    _Float16* featA = (_Float16*)(smem + 147456);     // 8*32*16 halfs  =  8KB (A frags, layer0)
    _Float16* actA  = (_Float16*)(smem + 155648);     // 4*32*16 halfs  =  4KB (A frags, layer1)
    float*    act2  = (float*)(smem + 159744);        // 16*64 f32      =  4KB
    // total dynamic LDS: 163840 B (160KB of 320KB/WGP)

    const int tid  = threadIdx.x;
    const int lane = tid & 31;
    const int wv   = tid >> 5;
    const int b0i  = blockIdx.x * PAIRS;

    // ---- Phase 1a: copy staged f16 weights (80KB) from L2-resident d_ws ----
    {
        const uint4* src = (const uint4*)wsrc;
        uint4*       dst = (uint4*)Wst;
        for (int i = tid; i < (int)(WST_HALFS * 2 / 16); i += THREADS) dst[i] = src[i];
    }
    // ---- Phase 1b: gather 16 pairs x 2 sides x S rows of K f32 ----
    for (int i = tid; i < PAIRS * 2 * S_ * (K_ / 4); i += THREADS) {
        int k4 = i & 31, s = (i >> 5) & 3, side = (i >> 7) & 1, p = i >> 8;
        int nd = side ? idx1[b0i + p] : idx0[b0i + p];
        const float* srcb = side ? Y : X;
        float4 v = *(const float4*)(srcb + ((size_t)s * N_ + nd) * K_ + k4 * 4);
        *(float4*)(Zf + ((p * 2 + side) * S_ + s) * K_ + k4 * 4) = v;
    }
    __syncthreads();

    // ---- Phase 2: embeddings; 32 (pair,side) tasks over 4 waves.
    //      Results written as f16 directly in A-fragment order:
    //      element (M=p, K=k'): ks=k'>>5, kr=k'&31, hi=(kr>>3)&1,
    //      j=(kr&7)+((kr>>4)<<3), L=p+(hi<<4)  ->  featA[(ks*32+L)*16+j]
    for (int t = wv; t < PAIRS * 2; t += 4) {
        int side = t & 1, p = t >> 1;
        const float* Zb  = Zf + (p * 2 + side) * (S_ * K_);
        const float* Wc1 = side ? WcY1 : WcX1;
        const float* Wc2 = side ? WcY2 : WcX2;
        const float* bc2 = side ? bcY2 : bcX2;
        const float  bc1 = side ? bcY1[0] : bcX1[0];
        const float* av  = side ? yatt : xatt;

        float c1 = 0.f;
        #pragma unroll
        for (int j = 0; j < 16; ++j) { int m = j * 32 + lane; c1 += Zb[m] * Wc1[m]; }
        #pragma unroll
        for (int o = 16; o; o >>= 1) c1 += __shfl_xor(c1, o, 32);
        c1 = lrelu(c1 + bc1);

        float tv[4], mx = -3.4e38f;
        #pragma unroll
        for (int j = 0; j < 4; ++j) { int k = lane + 32 * j; tv[j] = c1 * Wc2[k] + bc2[k]; mx = fmaxf(mx, tv[j]); }
        #pragma unroll
        for (int o = 16; o; o >>= 1) mx = fmaxf(mx, __shfl_xor(mx, o, 32));
        float ssum = 0.f;
        #pragma unroll
        for (int j = 0; j < 4; ++j) { tv[j] = __expf(tv[j] - mx); ssum += tv[j]; }
        #pragma unroll
        for (int o = 16; o; o >>= 1) ssum += __shfl_xor(ssum, o, 32);
        float inv = 1.f / ssum;

        float a0 = av[0], a1 = av[1], a2 = av[2], a3 = av[3];
        float m4 = fmaxf(fmaxf(a0, a1), fmaxf(a2, a3));
        float e0 = __expf(a0 - m4), e1 = __expf(a1 - m4), e2 = __expf(a2 - m4), e3 = __expf(a3 - m4);
        float is4 = 1.f / (e0 + e1 + e2 + e3);
        float ws[4] = { e0 * is4, e1 * is4, e2 * is4, e3 * is4 };

        #pragma unroll
        for (int j = 0; j < 4; ++j) {
            int k  = lane + 32 * j;
            float zs = 0.f, e8 = 0.f;
            #pragma unroll
            for (int s = 0; s < S_; ++s) { float z = Zb[s * K_ + k]; zs += z; e8 += ws[s] * z; }
            float val = 0.5f * (zs * (tv[j] * inv) + e8);
            int kp = side * K_ + k;                       // 0..255
            int ks = kp >> 5, kr = kp & 31;
            int hi2 = (kr >> 3) & 1, jj = (kr & 7) + ((kr >> 4) << 3);
            featA[((ks * 32) + p + (hi2 << 4)) * 16 + jj] = (_Float16)val;
        }
    }
    __syncthreads();

    const int hi = lane >> 4;
    const int cm = lane & 15;

    // ---- Layer 0: [16,256] @ [256,128], 8 k-steps, 2 N-tiles per wave ----
    {
        CFrag acc0, acc1;
        #pragma unroll
        for (int r = 0; r < 8; ++r) { acc0.f[r] = 0.f; acc1.f[r] = 0.f; }
        const int nt0 = wv, nt1 = wv + 4;
        #pragma unroll
        for (int ks = 0; ks < 8; ++ks) {
            v16h a   = *(const v16h*)(featA + (ks * 32 + lane) * 16);
            v16h bf0 = *(const v16h*)(Wst + ((ks * 8 + nt0) * 32 + lane) * 16);
            v16h bf1 = *(const v16h*)(Wst + ((ks * 8 + nt1) * 32 + lane) * 16);
            acc0.v = __builtin_amdgcn_wmma_f32_16x16x32_f16(false, a, false, bf0, (short)0, acc0.v, false, false);
            acc1.v = __builtin_amdgcn_wmma_f32_16x16x32_f16(false, a, false, bf1, (short)0, acc1.v, false, false);
        }
        // bias + lrelu, write as f16 A-fragments for layer 1
        const int n0 = nt0 * 16 + cm, n1 = nt1 * 16 + cm;
        const float bb0 = b0[n0], bb1 = b0[n1];
        #pragma unroll
        for (int r = 0; r < 8; ++r) {
            int row = r + hi * 8;
            {
                float v = lrelu(acc0.f[r] + bb0);
                int ks = n0 >> 5, kr = n0 & 31;
                int h2 = (kr >> 3) & 1, jj = (kr & 7) + ((kr >> 4) << 3);
                actA[((ks * 32) + row + (h2 << 4)) * 16 + jj] = (_Float16)v;
            }
            {
                float v = lrelu(acc1.f[r] + bb1);
                int ks = n1 >> 5, kr = n1 & 31;
                int h2 = (kr >> 3) & 1, jj = (kr & 7) + ((kr >> 4) << 3);
                actA[((ks * 32) + row + (h2 << 4)) * 16 + jj] = (_Float16)v;
            }
        }
    }
    __syncthreads();

    // ---- Layer 1: [16,128] @ [128,64], 4 k-steps, 1 N-tile per wave ----
    {
        CFrag acc;
        #pragma unroll
        for (int r = 0; r < 8; ++r) acc.f[r] = 0.f;
        const _Float16* W1s = Wst + W0_ELEMS;
        #pragma unroll
        for (int ks = 0; ks < 4; ++ks) {
            v16h a  = *(const v16h*)(actA + (ks * 32 + lane) * 16);
            v16h bf = *(const v16h*)(W1s + ((ks * 4 + wv) * 32 + lane) * 16);
            acc.v = __builtin_amdgcn_wmma_f32_16x16x32_f16(false, a, false, bf, (short)0, acc.v, false, false);
        }
        const int n = wv * 16 + cm;
        const float bb = b1[n];
        #pragma unroll
        for (int r = 0; r < 8; ++r) {
            int row = r + hi * 8;
            act2[row * 64 + n] = lrelu(acc.f[r] + bb);
        }
    }
    __syncthreads();

    // ---- Layer 2: [16,64] @ [64,1] + sigmoid (wave dot-product) ----
    {
        const float w2a = W2[lane], w2b = W2[lane + 32], b2v = b2[0];
        #pragma unroll
        for (int pp = 0; pp < 4; ++pp) {
            int p = wv * 4 + pp;
            float s = act2[p * 64 + lane] * w2a + act2[p * 64 + lane + 32] * w2b;
            #pragma unroll
            for (int o = 16; o; o >>= 1) s += __shfl_xor(s, o, 32);
            if (lane == 0) out[b0i + p] = 1.f / (1.f + __expf(-(s + b2v)));
        }
    }
}

extern "C" void kernel_launch(void* const* d_in, const int* in_sizes, int n_in,
                              void* d_out, int out_size, void* d_ws, size_t ws_size,
                              hipStream_t stream) {
    const float* X    = (const float*)d_in[0];
    const float* Y    = (const float*)d_in[1];
    const float* xatt = (const float*)d_in[2];
    const float* yatt = (const float*)d_in[3];
    const float* WcX1 = (const float*)d_in[4];
    const float* bcX1 = (const float*)d_in[5];
    const float* WcX2 = (const float*)d_in[6];
    const float* bcX2 = (const float*)d_in[7];
    const float* WcY1 = (const float*)d_in[8];
    const float* bcY1 = (const float*)d_in[9];
    const float* WcY2 = (const float*)d_in[10];
    const float* bcY2 = (const float*)d_in[11];
    const float* W0   = (const float*)d_in[12];
    const float* b0   = (const float*)d_in[13];
    const float* W1   = (const float*)d_in[14];
    const float* b1   = (const float*)d_in[15];
    const float* W2   = (const float*)d_in[16];
    const float* b2   = (const float*)d_in[17];
    const int*   idx0 = (const int*)d_in[18];
    const int*   idx1 = (const int*)d_in[19];

    _Float16* wst = (_Float16*)d_ws;              // 81920 B staged weights

    // One-shot weight staging (cheap: 40960 elements)
    MF_stage_weights<<<(WST_HALFS + 255) / 256, 256, 0, stream>>>(W0, W1, wst);

    const int B = in_sizes[18];
    const int blocks = B / PAIRS;
    const size_t smem = 163840;                   // 160KB dynamic LDS

    (void)hipFuncSetAttribute((const void*)MF_pairmlp_wmma,
                              hipFuncAttributeMaxDynamicSharedMemorySize, (int)smem);

    MF_pairmlp_wmma<<<blocks, THREADS, smem, stream>>>(
        X, Y, xatt, yatt, WcX1, bcX1, WcX2, bcX2, WcY1, bcY1, WcY2, bcY2,
        b0, b1, W2, b2, wst, idx0, idx1, (float*)d_out);
}
